// Model_82300163326283
// MI455X (gfx1250) — matrix operand order repaired and checked by reading, not compiled, not run on hardware
//
#include <hip/hip_runtime.h>
#include <hip/hip_bf16.h>

// ---------------------------------------------------------------------------
// Types for CDNA5 WMMA (wave32, 16x16x32 bf16 -> f32)
// ---------------------------------------------------------------------------
typedef __bf16 bf16_t;
typedef __attribute__((ext_vector_type(16))) __bf16 v16bf;
typedef __attribute__((ext_vector_type(8)))  __bf16 v8bf;
typedef __attribute__((ext_vector_type(8)))  float  v8f;
typedef int v4i_vs __attribute__((vector_size(16)));   // matches builtin param type

// CDNA5 async global->LDS copy path (ASYNCcnt), with safe fallback.
#if defined(__gfx1250__) && __has_builtin(__builtin_amdgcn_global_load_async_to_lds_b128) && \
    __has_builtin(__builtin_amdgcn_s_wait_asynccnt)
#define HAVE_ASYNC_LDS 1
#else
#define HAVE_ASYNC_LDS 0
#endif

__device__ __forceinline__ bf16_t f2bf(float f) {
  union { float f; unsigned u; } x; x.f = f;
  unsigned r = (x.u + 0x7FFFu + ((x.u >> 16) & 1u)) >> 16;   // RNE
  union { unsigned short s; bf16_t b; } y; y.s = (unsigned short)r;
  return y.b;
}

// ---------------------------------------------------------------------------
// Utility kernels
// ---------------------------------------------------------------------------
__global__ void k_zero(float* __restrict__ p, long n) {
  long i = (long)blockIdx.x * blockDim.x + threadIdx.x;
  if (i < n) p[i] = 0.0f;
}

__global__ void k_f32_to_bf16(const float* __restrict__ src, bf16_t* __restrict__ dst, long n) {
  long i = (long)blockIdx.x * blockDim.x + threadIdx.x;
  if (i < n) dst[i] = f2bf(src[i]);
}

// ---------------------------------------------------------------------------
// Edge scatter: one wave per edge, 128 floats = 32 lanes x float4.
// Native fp32 atomics (global_atomic_add_f32) via unsafeAtomicAdd.
// ---------------------------------------------------------------------------
__global__ __launch_bounds__(256)
void k_scatter(const float* __restrict__ x, const int* __restrict__ ei,
               float* __restrict__ msg, float* __restrict__ cnt, int E) {
  int e = blockIdx.x * 8 + (threadIdx.x >> 5);
  if (e >= E) return;
  int lane = threadIdx.x & 31;
  int src = ei[e];
  int dst = ei[(size_t)E + e];
  float4 v = ((const float4*)(x + (size_t)src * 128))[lane];
  float* m = msg + (size_t)dst * 128 + lane * 4;
  unsafeAtomicAdd(m + 0, v.x);
  unsafeAtomicAdd(m + 1, v.y);
  unsafeAtomicAdd(m + 2, v.z);
  unsafeAtomicAdd(m + 3, v.w);
  if (lane == 0) unsafeAtomicAdd(&cnt[dst], 1.0f);
}

// aggr = msg / max(cnt,1) -> bf16 ; x -> bf16 ; x -> target output copy
__global__ void k_aggr_pack(const float* __restrict__ msg, const float* __restrict__ cnt,
                            const float* __restrict__ x,
                            bf16_t* __restrict__ Abf, bf16_t* __restrict__ Xbf,
                            float* __restrict__ target, long n128) {
  long i = (long)blockIdx.x * blockDim.x + threadIdx.x;
  if (i >= n128) return;
  long node = i >> 7;
  float c = cnt[node];
  c = c > 1.0f ? c : 1.0f;
  Abf[i] = f2bf(msg[i] / c);
  float xv = x[i];
  Xbf[i] = f2bf(xv);
  target[i] = xv;
}

// ---------------------------------------------------------------------------
// Fragment loaders
// A (16x32, MxK): lane = m + 16h; VGPR0-3 <- K[k0+8h .. +8), VGPR4-7 <- K[k0+16+8h .. +8)
// B (32x16, KxN) from W[N,K] row-major (B[k,n] = W[n,k]):
//   lane = n + 16h; VGPR0-3 <- K[k0+8h .. +8), VGPR4-7 <- K[k0+16+8h .. +8)
// ---------------------------------------------------------------------------
__device__ __forceinline__ v16bf load_fragA_lds(const bf16_t* lds, int strideE,
                                                int m, int k0, int hsel) {
  union { v16bf v; v8bf h[2]; } f;
  const bf16_t* base = lds + m * strideE + k0 + 8 * hsel;
  f.h[0] = *(const v8bf*)(base);        // ds_load_b128
  f.h[1] = *(const v8bf*)(base + 16);   // ds_load_b128
  return f.v;
}

__device__ __forceinline__ v16bf load_fragB(const bf16_t* __restrict__ W, int K,
                                            int col, int k0, int hsel) {
  union { v16bf v; v8bf h[2]; } f;
  const bf16_t* base = W + (size_t)col * K + k0 + 8 * hsel;
  f.h[0] = *(const v8bf*)(base);
  f.h[1] = *(const v8bf*)(base + 16);
  return f.v;
}

// ---------------------------------------------------------------------------
// WMMA GEMM: out = act( A @ W^T [+ A2 @ W2^T] + bias )
//   act(v) = v >= 0 ? v : slope*v (slope from device ptr if given; 1.0 = identity)
// Block = 8 waves; all waves share one 16-row A tile staged in LDS via the
// CDNA5 async global->LDS copy; each wave owns one 16-col N tile; B fragments
// are software-pipelined from global (weights are L2-resident).
// K is compile-time (128 or 512) so the chunk loop fully unrolls: the
// prefetch rotation becomes register renaming, no branches in the hot loop.
// grid.x = M/16 ; grid.y = Nout/128 ; dyn-LDS = (DUAL?2:1)*16*(K+8)*2 bytes
// ---------------------------------------------------------------------------
template <int DUAL, int K>
__global__ __launch_bounds__(256)
void k_gemm_wmma(const bf16_t* __restrict__ A,  const bf16_t* __restrict__ W,
                 const bf16_t* __restrict__ A2, const bf16_t* __restrict__ W2,
                 const float* __restrict__ bias, const float* __restrict__ slope_ptr,
                 float slope_const,
                 float* __restrict__ outF, bf16_t* __restrict__ outBF,
                 int M, int Nout) {
  extern __shared__ __align__(16) char smem_raw[];
  bf16_t* ldsA  = (bf16_t*)smem_raw;
  constexpr int KP = K + 8;              // padded row stride: lane m -> word 4m mod 64
  bf16_t* ldsA2 = ldsA + 16 * KP;

  const int tid = threadIdx.x;
  const int m0  = blockIdx.x * 16;

  // ---- cooperative staging of the A tile(s) into LDS, 16-byte units ----
  {
    constexpr int upr   = K >> 3;        // 16B units per row
    constexpr int units = 16 * upr;      // units per matrix
    constexpr int total = units * (DUAL ? 2 : 1);
#pragma unroll
    for (int u0 = 0; u0 < total; u0 += 256) {
      const int u = u0 + tid;
      if (total % 256 == 0 || u < total) {
        const int which = DUAL ? (u / units) : 0;
        const int uu    = u - which * units;
        const int row   = uu / upr;
        const int c16   = uu - row * upr;
        int grow = m0 + row; if (grow >= M) grow = M - 1;
        const bf16_t* gsrc = (which ? A2 : A) + ((size_t)grow * K + c16 * 8);
        bf16_t*       ldst = (which ? ldsA2 : ldsA) + (row * KP + c16 * 8);
#if HAVE_ASYNC_LDS
        __builtin_amdgcn_global_load_async_to_lds_b128(
            (__attribute__((address_space(1))) v4i_vs*)gsrc,
            (__attribute__((address_space(3))) v4i_vs*)ldst, 0, 0);
#else
        *(uint4*)ldst = *(const uint4*)gsrc;
#endif
      }
    }
  }
#if HAVE_ASYNC_LDS
  __builtin_amdgcn_s_wait_asynccnt(0);
#endif
  __syncthreads();

  const int wave = tid >> 5;
  const int n0   = (wave + 8 * blockIdx.y) * 16;
  const int lane = tid & 31;
  const int hsel = lane >> 4;
  const int mn   = lane & 15;
  const int ncol = n0 + mn;

  constexpr int nchunks = K >> 5;
  v8f acc = {};

  // software pipeline: B fragments for chunk c+1 in flight during chunk c's WMMA
  v16bf b  = load_fragB(W, K, ncol, 0, hsel);
  v16bf b2 = {};
  if (DUAL) b2 = load_fragB(W2, K, ncol, 0, hsel);

#pragma unroll
  for (int c = 0; c < nchunks; ++c) {
    v16bf bc  = b;
    v16bf b2c = b2;
    if (c + 1 < nchunks) {
      b = load_fragB(W, K, ncol, (c + 1) * 32, hsel);
      if (DUAL) b2 = load_fragB(W2, K, ncol, (c + 1) * 32, hsel);
    }
    v16bf a = load_fragA_lds(ldsA, KP, mn, c * 32, hsel);
    acc = __builtin_amdgcn_wmma_f32_16x16x32_bf16(false, a, false, bc,
                                                  (short)0, acc, false, false);
    if (DUAL) {
      v16bf a2 = load_fragA_lds(ldsA2, KP, mn, c * 32, hsel);
      acc = __builtin_amdgcn_wmma_f32_16x16x32_bf16(false, a2, false, b2c,
                                                    (short)0, acc, false, false);
    }
  }

  const float slope = slope_ptr ? slope_ptr[0] : slope_const;
  const int col = n0 + mn;
  const float bv = bias ? bias[col] : 0.0f;

#pragma unroll
  for (int r = 0; r < 8; ++r) {
    int row = m0 + r + 8 * hsel;          // C/D layout: VGPR r -> M = r + 8*(lane/16)
    if (row < M) {
      float v = acc[r] + bv;
      v = v >= 0.0f ? v : slope * v;
      if (outF)  outF[(size_t)row * Nout + col] = v;
      if (outBF) outBF[(size_t)row * Nout + col] = f2bf(v);
    }
  }
}

// ---------------------------------------------------------------------------
// BatchNorm statistics: per-column partial sums over 512-row stripes,
// combined with fp32 global atomics.  block=256, grid=(ceil(N/512), 2)
// ---------------------------------------------------------------------------
__global__ __launch_bounds__(256)
void k_bn_stats(const float* __restrict__ d, float* __restrict__ colsum,
                float* __restrict__ colsumsq, int Nrows) {
  int col = threadIdx.x + 256 * blockIdx.y;            // 0..511
  int r0 = blockIdx.x * 512;
  int rend = r0 + 512; if (rend > Nrows) rend = Nrows;
  float s = 0.0f, s2 = 0.0f;
  for (int r = r0; r < rend; ++r) {
    float v = d[(size_t)r * 512 + col];
    s += v;
    s2 = fmaf(v, v, s2);
  }
  unsafeAtomicAdd(&colsum[col], s);
  unsafeAtomicAdd(&colsumsq[col], s2);
}

__global__ void k_bn_finalize(const float* __restrict__ colsum, const float* __restrict__ colsumsq,
                              const float* __restrict__ gamma, const float* __restrict__ beta,
                              float* __restrict__ scale, float* __restrict__ shift, float invN) {
  int c = threadIdx.x;                                  // 512 threads
  float mu  = colsum[c] * invN;
  float var = colsumsq[c] * invN - mu * mu;
  float rs  = rsqrtf(var + 1e-5f);
  float sc  = gamma[c] * rs;
  scale[c] = sc;
  shift[c] = beta[c] - mu * sc;
}

// dnorm = leaky01(d*scale[col]+shift[col]) -> bf16
__global__ void k_bn_apply(const float* __restrict__ d, const float* __restrict__ scale,
                           const float* __restrict__ shift, bf16_t* __restrict__ Dbf, long n) {
  long i = (long)blockIdx.x * blockDim.x + threadIdx.x;
  if (i >= n) return;
  int col = (int)(i & 511);
  float v = fmaf(d[i], scale[col], shift[col]);
  v = v >= 0.0f ? v : 0.01f * v;
  Dbf[i] = f2bf(v);
}

// ---------------------------------------------------------------------------
// Host launcher
// ---------------------------------------------------------------------------
extern "C" void kernel_launch(void* const* d_in, const int* in_sizes, int n_in,
                              void* d_out, int out_size, void* d_ws, size_t ws_size,
                              hipStream_t stream) {
  const float* x       = (const float*)d_in[0];
  const int*   ei      = (const int*)  d_in[1];
  const float* Wl      = (const float*)d_in[2];
  const float* bl      = (const float*)d_in[3];
  const float* Wr      = (const float*)d_in[4];
  const float* Wp      = (const float*)d_in[5];
  const float* bp      = (const float*)d_in[6];
  const float* a_prelu = (const float*)d_in[7];
  const float* Wd1     = (const float*)d_in[8];
  const float* bd1     = (const float*)d_in[9];
  const float* gamma   = (const float*)d_in[10];
  const float* beta    = (const float*)d_in[11];
  const float* Wd2     = (const float*)d_in[12];
  const float* bd2     = (const float*)d_in[13];

  const long N = in_sizes[0] / 128;     // 100000 (multiple of 16)
  const long E = in_sizes[1] / 2;       // 1600000
  if (N <= 0) return;

  // ---- workspace layout (bytes). The N x 512 f32 'd' buffer reuses the
  // region of msg/cnt/Abf/Xbf, which are dead before it is written. ----
  char* ws = (char*)d_ws;
  const size_t off_d   = 0;                                               // N*512*4
  const size_t off_msg = 0;                                               // N*128*4
  const size_t off_cnt = off_msg + (size_t)N * 128 * 4;                   // N*4
  const size_t off_Abf = (off_cnt + (size_t)N * 4 + 255) & ~(size_t)255;  // N*128*2
  const size_t off_Xbf = off_Abf + (size_t)N * 128 * 2;                   // N*128*2
  const size_t off_Hbf = off_d + (size_t)N * 512 * 4;                     // N*128*2
  const size_t off_Dbf = off_Hbf + (size_t)N * 128 * 2;                   // N*512*2
  const size_t off_Wl  = off_Dbf + (size_t)N * 512 * 2;
  const size_t off_Wr  = off_Wl  + 128 * 128 * 2;
  const size_t off_Wp  = off_Wr  + 128 * 128 * 2;
  const size_t off_Wd1 = off_Wp  + 128 * 128 * 2;
  const size_t off_Wd2 = off_Wd1 + 512 * 128 * 2;
  const size_t off_cs  = off_Wd2 + 128 * 512 * 2;
  const size_t off_cs2 = off_cs  + 512 * 4;
  const size_t off_sc  = off_cs2 + 512 * 4;
  const size_t off_sh  = off_sc  + 512 * 4;

  float*  msg   = (float*) (ws + off_msg);
  float*  cnt   = (float*) (ws + off_cnt);
  bf16_t* Abf   = (bf16_t*)(ws + off_Abf);
  bf16_t* Xbf   = (bf16_t*)(ws + off_Xbf);
  float*  dbuf  = (float*) (ws + off_d);
  bf16_t* Hbf   = (bf16_t*)(ws + off_Hbf);
  bf16_t* Dbf   = (bf16_t*)(ws + off_Dbf);
  bf16_t* Wlbf  = (bf16_t*)(ws + off_Wl);
  bf16_t* Wrbf  = (bf16_t*)(ws + off_Wr);
  bf16_t* Wpbf  = (bf16_t*)(ws + off_Wp);
  bf16_t* Wd1bf = (bf16_t*)(ws + off_Wd1);
  bf16_t* Wd2bf = (bf16_t*)(ws + off_Wd2);
  float*  csum  = (float*) (ws + off_cs);
  float*  csum2 = (float*) (ws + off_cs2);
  float*  scl   = (float*) (ws + off_sc);
  float*  shf   = (float*) (ws + off_sh);

  float* out_h   = (float*)d_out;
  float* out_z   = out_h   + (size_t)N * 128;
  float* out_rec = out_z   + (size_t)N * 128;
  float* out_tgt = out_rec + (size_t)N * 128;

  const dim3 blk(256);
  auto blocks = [](long n, int b) { return (unsigned)((n + b - 1) / b); };

  // 1) zero accumulators (msg+cnt contiguous) and BN stats
  k_zero<<<blocks(N * 129, 256), blk, 0, stream>>>(msg, N * 129);
  k_zero<<<blocks(1024, 256), blk, 0, stream>>>(csum, 1024);

  // 2) weights -> bf16
  k_f32_to_bf16<<<blocks(128 * 128, 256), blk, 0, stream>>>(Wl,  Wlbf,  128 * 128);
  k_f32_to_bf16<<<blocks(128 * 128, 256), blk, 0, stream>>>(Wr,  Wrbf,  128 * 128);
  k_f32_to_bf16<<<blocks(128 * 128, 256), blk, 0, stream>>>(Wp,  Wpbf,  128 * 128);
  k_f32_to_bf16<<<blocks(512 * 128, 256), blk, 0, stream>>>(Wd1, Wd1bf, 512 * 128);
  k_f32_to_bf16<<<blocks(128 * 512, 256), blk, 0, stream>>>(Wd2, Wd2bf, 128 * 512);

  // 3) edge scatter (segment sums + degree counts)
  k_scatter<<<blocks(E, 8), blk, 0, stream>>>(x, ei, msg, cnt, (int)E);

  // 4) mean-aggregate -> bf16; pack x -> bf16; copy target
  k_aggr_pack<<<blocks(N * 128, 256), blk, 0, stream>>>(msg, cnt, x, Abf, Xbf, out_tgt, N * 128);

  const unsigned mt = (unsigned)((N + 15) / 16);
  const size_t smemK128  = 16 * (128 + 8) * 2;   // 4352 B
  const size_t smemDual  = 2 * smemK128;         // 8704 B
  const size_t smemK512  = 16 * (512 + 8) * 2;   // 16640 B

  // 5) encoder: h = leaky_0.5( aggr@Wl^T + x@Wr^T + bl ) -> out_h (f32) + Hbf
  k_gemm_wmma<1, 128><<<dim3(mt, 1), blk, smemDual, stream>>>(
      Abf, Wlbf, Xbf, Wrbf, bl, nullptr, 0.5f, out_h, Hbf, (int)N, 128);

  // 6) projection: z = PReLU( h@Wp^T + bp, a ) -> out_z
  k_gemm_wmma<0, 128><<<dim3(mt, 1), blk, smemK128, stream>>>(
      Hbf, Wpbf, nullptr, nullptr, bp, a_prelu, 0.25f, out_z, nullptr, (int)N, 128);

  // 7) decoder stage 1: d = h@Wd1^T + bd1 (identity) -> dbuf (f32, N x 512)
  k_gemm_wmma<0, 128><<<dim3(mt, 4), blk, smemK128, stream>>>(
      Hbf, Wd1bf, nullptr, nullptr, bd1, nullptr, 1.0f, dbuf, nullptr, (int)N, 512);

  // 8) BatchNorm: column stats -> affine params -> apply + leaky_0.01 -> bf16
  k_bn_stats<<<dim3(blocks(N, 512), 2), blk, 0, stream>>>(dbuf, csum, csum2, (int)N);
  k_bn_finalize<<<1, 512, 0, stream>>>(csum, csum2, gamma, beta, scl, shf, 1.0f / (float)N);
  k_bn_apply<<<blocks(N * 512, 256), blk, 0, stream>>>(dbuf, scl, shf, Dbf, N * 512);

  // 9) decoder stage 2: rec = dnorm@Wd2^T + bd2 -> out_rec
  k_gemm_wmma<0, 512><<<dim3(mt, 1), blk, smemK512, stream>>>(
      Dbf, Wd2bf, nullptr, nullptr, bd2, nullptr, 1.0f, out_rec, nullptr, (int)N, 128);
}
